// Main_2568390443405
// MI455X (gfx1250) — matrix operand order repaired and checked by reading, not compiled, not run on hardware
//
#include <hip/hip_runtime.h>
#include <hip/hip_bf16.h>
#include <math.h>

typedef __attribute__((ext_vector_type(16))) _Float16 v16h;
typedef __attribute__((ext_vector_type(8)))  _Float16 v8h;
typedef __attribute__((ext_vector_type(4)))  _Float16 v4h;
typedef __attribute__((ext_vector_type(8)))  float    v8f;
typedef __attribute__((ext_vector_type(2)))  int      v2i;

// ---------------- model dims ----------------
#define MB 4
#define MT 1024
#define MD 512
#define MH 8
#define MDH 64
#define MLAYERS 4
#define MV 32000
#define MFF 2048
#define MM (MB * MT)   // 4096 token rows

// ---------------- gfx1250 async-to-LDS path (guarded) ----------------
#if defined(__HIP_DEVICE_COMPILE__) &&                                  \
    __has_builtin(__builtin_amdgcn_global_load_async_to_lds_b64) &&     \
    __has_builtin(__builtin_amdgcn_s_wait_asynccnt)
#define HAVE_ASYNC_LDS 1
typedef __attribute__((address_space(1))) v2i* gptr_v2i;   // int2 in global AS
typedef __attribute__((address_space(3))) v2i* lptr_v2i;   // int2 in LDS AS
#else
#define HAVE_ASYNC_LDS 0
#endif

// Stage one 8-byte chunk from global to LDS (async when available).
static __device__ __forceinline__ void stage_chunk(const _Float16* src,
                                                   _Float16* dst_lds) {
#if HAVE_ASYNC_LDS
  __builtin_amdgcn_global_load_async_to_lds_b64(
      (gptr_v2i)src, (lptr_v2i)dst_lds, 0, 0);
#else
  *(v4h*)dst_lds = *(const v4h*)src;
#endif
}
static __device__ __forceinline__ void wait_stage_prev() {  // <=1 outstanding
#if HAVE_ASYNC_LDS
  __builtin_amdgcn_s_wait_asynccnt(1);
#endif
}
static __device__ __forceinline__ void wait_stage_all() {   // drain
#if HAVE_ASYNC_LDS
  __builtin_amdgcn_s_wait_asynccnt(0);
#endif
}

static __device__ __forceinline__ v16h cat8(v8h lo, v8h hi) {
  return __builtin_shufflevector(lo, hi, 0,1,2,3,4,5,6,7,8,9,10,11,12,13,14,15);
}

// =====================================================================
// Pack fp32 weight [K,N] into f16 fragment-native layout:
//   Wp[((k>>5)*N + n)*32 + (k&31)]
// =====================================================================
__global__ void pack_w_kernel(const float* __restrict__ W,
                              _Float16* __restrict__ Wp, int K, int N) {
  const size_t total = (size_t)K * N;
  for (size_t e = (size_t)blockIdx.x * blockDim.x + threadIdx.x; e < total;
       e += (size_t)gridDim.x * blockDim.x) {
    const int kk = (int)(e / N), nn = (int)(e % N);
    Wp[((size_t)(kk >> 5) * N + nn) * 32 + (kk & 31)] = (_Float16)W[e];
  }
}

// =====================================================================
// Embedding: x = tok_emb[tokens] + pos_emb   (fp32 residual + f16 copy)
// =====================================================================
__global__ void embed_kernel(const int* __restrict__ tokens,
                             const float* __restrict__ tok_emb,
                             const float* __restrict__ pos_emb,
                             float* __restrict__ x, _Float16* __restrict__ xh) {
  const int row = blockIdx.x;
  const int t   = row % MT;
  const int tok = tokens[row];
  const float* te = tok_emb + (size_t)tok * MD;
  const float* pe = pos_emb + (size_t)t * MD;
  const size_t ro = (size_t)row * MD;
  for (int d = threadIdx.x; d < MD; d += blockDim.x) {
    const float v = te[d] + pe[d];
    x[ro + d] = v;
    xh[ro + d] = (_Float16)v;
  }
}

// =====================================================================
// WMMA GEMM:  C[M,N] = act(A[M,K] @ W[K,N] + bias[N])
//   Block = 8 waves = 32-row strip x (up to) 512 cols.
//   A strip (32x32 f16 per k-step) staged to LDS once per block via
//   async-to-LDS, double-buffered; next stage overlaps compute and is
//   released with s_wait_asynccnt 1 (async loads complete in-order).
//   Per wave: 32x64 tile = 2 A-frags x 4 B-frags = 8 WMMAs / k-step.
//   ACT: 0 none, 1 exact GELU
//   OMODE: 0 fp32 -> Cf, 1 f16 -> Ch, 2 f16 [B,H,DH,T] -> Ch,
//          3 fp32 nontemporal -> Cf (LM head stream-out)
// =====================================================================
template <int ACT, int OMODE>
__global__ __launch_bounds__(256)
void gemm_wmma_kernel(const _Float16* __restrict__ A,
                      const _Float16* __restrict__ Wp,
                      const float* __restrict__ bias,
                      float* __restrict__ Cf,
                      _Float16* __restrict__ Ch,
                      int M, int N, int K) {
  const int tid = threadIdx.x;
  const int lane = tid & 31, wid = tid >> 5;
  const int ntiles = N >> 6;
  const int NG = (ntiles + 7) >> 3;
  const int mt  = blockIdx.x / NG;
  const int nt  = (blockIdx.x % NG) * 8 + wid;
  const bool active = (nt < ntiles);
  const int lm = lane & 15, grp = lane >> 4;

  __shared__ __align__(16) _Float16 Asmem[2][32 * 32];  // double-buffered strip

  // per-thread 8B staging chunk: row sr, half-offset sc
  const int sr = tid >> 3, sc = (tid & 7) * 4;
  const _Float16* Asrc = A + (size_t)(mt * 32 + sr) * K + sc;
  _Float16* Adst0 = &Asmem[0][sr * 32 + sc];
  _Float16* Adst1 = &Asmem[1][sr * 32 + sc];

  v8f acc[4][2];
  #pragma unroll
  for (int i = 0; i < 4; ++i)
    #pragma unroll
    for (int m = 0; m < 2; ++m) acc[i][m] = v8f{};

  stage_chunk(Asrc, Adst0);             // prologue: stage kb = 0

  int cur = 0;
  for (int kb = 0; kb < K; kb += 32, cur ^= 1) {
    const bool has_next = (kb + 32) < K;
    if (has_next) {                     // stage next k-block into other buffer
      stage_chunk(Asrc + kb + 32, cur ? Adst0 : Adst1);
      wait_stage_prev();                // previous stage done, next in flight
    } else {
      wait_stage_all();                 // final buffer landed
    }
    __syncthreads();

    if (active) {
      const _Float16* Ab = &Asmem[cur][0];
      const v16h a0 = cat8(*(const v8h*)(Ab + lm * 32 + grp * 8),
                           *(const v8h*)(Ab + lm * 32 + grp * 8 + 16));
      const v16h a1 = cat8(*(const v8h*)(Ab + (16 + lm) * 32 + grp * 8),
                           *(const v8h*)(Ab + (16 + lm) * 32 + grp * 8 + 16));

      const _Float16* Bbase =
          Wp + ((size_t)(kb >> 5) * N + nt * 64) * 32 + grp * 8;
      if (has_next)                     // prefetch next weight k-block
        __builtin_prefetch((const void*)(Bbase + (size_t)N * 32), 0, 0);

      #pragma unroll
      for (int i = 0; i < 4; ++i) {
        const _Float16* bp = Bbase + (size_t)(i * 16 + lm) * 32;
        const v16h bf = cat8(*(const v8h*)bp, *(const v8h*)(bp + 16));
        acc[i][0] = __builtin_amdgcn_wmma_f32_16x16x32_f16(
            false, a0, false, bf, (short)0, acc[i][0], false, false);
        acc[i][1] = __builtin_amdgcn_wmma_f32_16x16x32_f16(
            false, a1, false, bf, (short)0, acc[i][1], false, false);
      }
    }
    __syncthreads();                    // reads done before buffer is restaged
  }

  if (!active) return;

  // ---- epilogue ----
  #pragma unroll
  for (int i = 0; i < 4; ++i) {
    const int col = nt * 64 + i * 16 + lm;
    const float bcol = bias[col];
    #pragma unroll
    for (int m = 0; m < 2; ++m) {
      #pragma unroll
      for (int r = 0; r < 8; ++r) {
        const int row = mt * 32 + m * 16 + grp * 8 + r;
        float v = acc[i][m][r] + bcol;
        if (ACT == 1) v = 0.5f * v * (1.0f + erff(v * 0.70710678118654752f));
        if (OMODE == 0) {
          Cf[(size_t)row * N + col] = v;
        } else if (OMODE == 1) {
          Ch[(size_t)row * N + col] = (_Float16)v;
        } else if (OMODE == 2) {        // per-head transposed V [B,H,DH,T]
          const int bidx = row >> 10, t = row & (MT - 1);
          const int hidx = col >> 6, dh = col & 63;
          Ch[(((size_t)bidx * MH + hidx) * MDH + dh) * MT + t] = (_Float16)v;
        } else {                        // OMODE 3: streaming fp32 (NT hint)
          __builtin_nontemporal_store(v, &Cf[(size_t)row * N + col]);
        }
      }
    }
  }
}

// =====================================================================
// Causal flash attention, all-f16 operands. One wave per (b,h,16 queries).
// q,k: f16 [B*T, D] row-major. vT: f16 [B,H,DH,T]. o: f16 [B*T, D].
// =====================================================================
__global__ __launch_bounds__(32)
void attn_kernel(const _Float16* __restrict__ q, const _Float16* __restrict__ k,
                 const _Float16* __restrict__ vT, _Float16* __restrict__ o) {
  const int lane = threadIdx.x;
  const int qt = blockIdx.x, h = blockIdx.y, b = blockIdx.z;
  const int lm = lane & 15, grp = lane >> 4;
  const float scale = 0.125f;           // 1/sqrt(64)
  const size_t rowbase = (size_t)b * MT;
  const size_t hoff = (size_t)h * MDH;

  v16h qa[2];
  const _Float16* Qp = q + (rowbase + qt * 16 + lm) * MD + hoff;
  #pragma unroll
  for (int s = 0; s < 2; ++s)
    qa[s] = cat8(*(const v8h*)(Qp + s * 32 + grp * 8),
                 *(const v8h*)(Qp + s * 32 + grp * 8 + 16));

  float Mx[8], Ls[8];
  #pragma unroll
  for (int r = 0; r < 8; ++r) { Mx[r] = -3.0e38f; Ls[r] = 0.0f; }
  v8f oacc[4];
  #pragma unroll
  for (int d = 0; d < 4; ++d) oacc[d] = v8f{};

  __shared__ __align__(16) _Float16 Pl[16 * 32];   // P relayout bounce

  const int qmax = qt * 16 + 15;
  for (int s0 = 0; s0 <= qmax; s0 += 32) {
    v8f sc[2];
    #pragma unroll
    for (int t = 0; t < 2; ++t) sc[t] = v8f{};
    #pragma unroll
    for (int t = 0; t < 2; ++t) {
      #pragma unroll
      for (int s = 0; s < 2; ++s) {
        const _Float16* Kp =
            k + (rowbase + s0 + t * 16 + lm) * MD + hoff + s * 32 + grp * 8;
        const v16h kf = cat8(*(const v8h*)Kp, *(const v8h*)(Kp + 16));
        sc[t] = __builtin_amdgcn_wmma_f32_16x16x32_f16(
            false, qa[s], false, kf, (short)0, sc[t], false, false);
      }
    }

    float pv[2][8];
    #pragma unroll
    for (int t = 0; t < 2; ++t) {
      const int key = s0 + t * 16 + lm;
      #pragma unroll
      for (int r = 0; r < 8; ++r) {
        const int qi = qt * 16 + grp * 8 + r;
        const float x = sc[t][r] * scale;
        pv[t][r] = (key <= qi) ? x : -1.0e9f;
      }
    }

    float alpha[8];
    #pragma unroll
    for (int r = 0; r < 8; ++r) {
      float x = fmaxf(pv[0][r], pv[1][r]);
      #pragma unroll
      for (int off = 1; off < 16; off <<= 1) x = fmaxf(x, __shfl_xor(x, off, 16));
      const float Mn = fmaxf(Mx[r], x);
      alpha[r] = __expf(Mx[r] - Mn);
      Mx[r] = Mn;
    }
    #pragma unroll
    for (int t = 0; t < 2; ++t)
      #pragma unroll
      for (int r = 0; r < 8; ++r) pv[t][r] = __expf(pv[t][r] - Mx[r]);
    #pragma unroll
    for (int r = 0; r < 8; ++r) {
      float x = pv[0][r] + pv[1][r];
      #pragma unroll
      for (int off = 1; off < 16; off <<= 1) x += __shfl_xor(x, off, 16);
      Ls[r] = Ls[r] * alpha[r] + x;
    }
    #pragma unroll
    for (int d = 0; d < 4; ++d)
      #pragma unroll
      for (int r = 0; r < 8; ++r) oacc[d][r] *= alpha[r];

    #pragma unroll
    for (int t = 0; t < 2; ++t)
      #pragma unroll
      for (int r = 0; r < 8; ++r)
        Pl[(grp * 8 + r) * 32 + t * 16 + lm] = (_Float16)pv[t][r];
    __syncthreads();
    const v16h pa = cat8(*(const v8h*)&Pl[lm * 32 + grp * 8],
                         *(const v8h*)&Pl[lm * 32 + grp * 8 + 16]);

    #pragma unroll
    for (int d = 0; d < 4; ++d) {
      const _Float16* Vp =
          vT + (((size_t)b * MH + h) * MDH + d * 16 + lm) * MT + s0 + grp * 8;
      const v16h vf = cat8(*(const v8h*)Vp, *(const v8h*)(Vp + 16));
      oacc[d] = __builtin_amdgcn_wmma_f32_16x16x32_f16(
          false, pa, false, vf, (short)0, oacc[d], false, false);
    }
    __syncthreads();
  }

  #pragma unroll
  for (int d = 0; d < 4; ++d)
    #pragma unroll
    for (int r = 0; r < 8; ++r) {
      const int qi = qt * 16 + grp * 8 + r;
      o[(rowbase + qi) * MD + hoff + d * 16 + lm] =
          (_Float16)(oacc[d][r] / Ls[r]);
    }
}

// =====================================================================
// out = LayerNorm(x + y) * g + be   (y may be null). fp32 + f16 outputs.
// =====================================================================
__global__ void add_ln_kernel(const float* __restrict__ x,
                              const float* __restrict__ y,
                              const float* __restrict__ g,
                              const float* __restrict__ be,
                              float* __restrict__ outf,
                              _Float16* __restrict__ outh) {
  const int row = blockIdx.x, tid = threadIdx.x;
  const size_t ro = (size_t)row * MD;
  float vals[4];
  float s = 0.0f;
  #pragma unroll
  for (int i = 0; i < 4; ++i) {
    const int d = tid + i * 128;
    float t = x[ro + d];
    if (y) t += y[ro + d];
    vals[i] = t; s += t;
  }
  __shared__ float red[128];
  red[tid] = s; __syncthreads();
  for (int st = 64; st > 0; st >>= 1) { if (tid < st) red[tid] += red[tid + st]; __syncthreads(); }
  const float mean = red[0] * (1.0f / MD);
  __syncthreads();
  float s2 = 0.0f;
  #pragma unroll
  for (int i = 0; i < 4; ++i) { const float dd = vals[i] - mean; s2 += dd * dd; }
  red[tid] = s2; __syncthreads();
  for (int st = 64; st > 0; st >>= 1) { if (tid < st) red[tid] += red[tid + st]; __syncthreads(); }
  const float inv = rsqrtf(red[0] * (1.0f / MD) + 1e-5f);
  #pragma unroll
  for (int i = 0; i < 4; ++i) {
    const int d = tid + i * 128;
    const float v = (vals[i] - mean) * inv * g[d] + be[d];
    outf[ro + d] = v;
    outh[ro + d] = (_Float16)v;
  }
}

// =====================================================================
// Host side
// =====================================================================
static void launch_gemm(int act, int omode, const _Float16* A, const _Float16* Wp,
                        const float* bias, float* Cf, _Float16* Ch,
                        int M, int N, int K, hipStream_t s) {
  const int ntiles = N / 64;
  const int NG = (ntiles + 7) / 8;
  dim3 block(256), grid((M / 32) * NG);
  if (act == 0 && omode == 0)
    gemm_wmma_kernel<0, 0><<<grid, block, 0, s>>>(A, Wp, bias, Cf, Ch, M, N, K);
  else if (act == 0 && omode == 1)
    gemm_wmma_kernel<0, 1><<<grid, block, 0, s>>>(A, Wp, bias, Cf, Ch, M, N, K);
  else if (act == 0 && omode == 2)
    gemm_wmma_kernel<0, 2><<<grid, block, 0, s>>>(A, Wp, bias, Cf, Ch, M, N, K);
  else if (act == 0 && omode == 3)
    gemm_wmma_kernel<0, 3><<<grid, block, 0, s>>>(A, Wp, bias, Cf, Ch, M, N, K);
  else
    gemm_wmma_kernel<1, 1><<<grid, block, 0, s>>>(A, Wp, bias, Cf, Ch, M, N, K);
}

static void launch_pack(const float* W, _Float16* Wp, int K, int N, hipStream_t s) {
  pack_w_kernel<<<dim3(2048), dim3(256), 0, s>>>(W, Wp, K, N);
}

extern "C" void kernel_launch(void* const* d_in, const int* in_sizes, int n_in,
                              void* d_out, int out_size, void* d_ws, size_t ws_size,
                              hipStream_t stream) {
  (void)in_sizes; (void)n_in; (void)out_size; (void)ws_size;
  const int*   tokens  = (const int*)  d_in[0];
  const float* tok_emb = (const float*)d_in[1];
  const float* pos_emb = (const float*)d_in[2];
  const float* Wq = (const float*)d_in[3],  *bq = (const float*)d_in[4];
  const float* Wk = (const float*)d_in[5],  *bk = (const float*)d_in[6];
  const float* Wv = (const float*)d_in[7],  *bv = (const float*)d_in[8];
  const float* Wo = (const float*)d_in[9],  *bo = (const float*)d_in[10];
  const float* W1 = (const float*)d_in[11], *b1 = (const float*)d_in[12];
  const float* W2 = (const float*)d_in[13], *b2 = (const float*)d_in[14];
  const float* g1 = (const float*)d_in[15], *be1 = (const float*)d_in[16];
  const float* g2 = (const float*)d_in[17], *be2 = (const float*)d_in[18];
  const float* gf = (const float*)d_in[19], *bef = (const float*)d_in[20];
  const float* Wl = (const float*)d_in[21], *bl  = (const float*)d_in[22];
  float* out = (float*)d_out;

  // ---- workspace bump allocator (16B aligned) ----
  char* wsb = (char*)d_ws;
  size_t off = 0;
  auto allocf = [&](size_t n) {
    off = (off + 15) & ~(size_t)15;
    float* p = (float*)(wsb + off); off += n * 4; return p;
  };
  auto alloch = [&](size_t n) {
    off = (off + 15) & ~(size_t)15;
    _Float16* p = (_Float16*)(wsb + off); off += n * 2; return p;
  };

  float* x0f = allocf((size_t)MM * MD);
  float* x1f = allocf((size_t)MM * MD);
  float* ybf = allocf((size_t)MM * MD);
  _Float16* x0h = alloch((size_t)MM * MD);
  _Float16* x1h = alloch((size_t)MM * MD);
  _Float16* qh  = alloch((size_t)MM * MD);
  _Float16* kh  = alloch((size_t)MM * MD);
  _Float16* vTh = alloch((size_t)MM * MD);   // [B,H,DH,T]
  _Float16* oh  = alloch((size_t)MM * MD);
  _Float16* hh  = alloch((size_t)MM * MFF);
  _Float16* xfh = alloch((size_t)MM * MD);
  _Float16* Wqp = alloch((size_t)MLAYERS * MD * MD);
  _Float16* Wkp = alloch((size_t)MLAYERS * MD * MD);
  _Float16* Wvp = alloch((size_t)MLAYERS * MD * MD);
  _Float16* Wop = alloch((size_t)MLAYERS * MD * MD);
  _Float16* W1p = alloch((size_t)MLAYERS * MD * MFF);
  _Float16* W2p = alloch((size_t)MLAYERS * MFF * MD);
  _Float16* Wlp = alloch((size_t)MD * MV);

  for (int l = 0; l < MLAYERS; ++l) {
    const size_t wo = (size_t)l * MD * MD;
    launch_pack(Wq + wo, Wqp + wo, MD, MD, stream);
    launch_pack(Wk + wo, Wkp + wo, MD, MD, stream);
    launch_pack(Wv + wo, Wvp + wo, MD, MD, stream);
    launch_pack(Wo + wo, Wop + wo, MD, MD, stream);
    launch_pack(W1 + (size_t)l * MD * MFF, W1p + (size_t)l * MD * MFF, MD, MFF, stream);
    launch_pack(W2 + (size_t)l * MFF * MD, W2p + (size_t)l * MFF * MD, MFF, MD, stream);
  }
  launch_pack(Wl, Wlp, MD, MV, stream);

  embed_kernel<<<dim3(MM), dim3(256), 0, stream>>>(tokens, tok_emb, pos_emb, x0f, x0h);

  const float* xcF = x0f;
  const _Float16* xcH = x0h;
  for (int l = 0; l < MLAYERS; ++l) {
    const size_t wo = (size_t)l * MD * MD, bo1 = (size_t)l * MD;
    const size_t w1o = (size_t)l * MD * MFF, b1o = (size_t)l * MFF;
    const size_t w2o = (size_t)l * MFF * MD;

    launch_gemm(0, 1, xcH, Wqp + wo, bq + bo1, nullptr, qh,  MM, MD, MD, stream);
    launch_gemm(0, 1, xcH, Wkp + wo, bk + bo1, nullptr, kh,  MM, MD, MD, stream);
    launch_gemm(0, 2, xcH, Wvp + wo, bv + bo1, nullptr, vTh, MM, MD, MD, stream);

    attn_kernel<<<dim3(MT / 16, MH, MB), dim3(32), 0, stream>>>(qh, kh, vTh, oh);

    launch_gemm(0, 0, oh, Wop + wo, bo + bo1, ybf, nullptr, MM, MD, MD, stream);
    add_ln_kernel<<<dim3(MM), dim3(128), 0, stream>>>(xcF, ybf, g1 + bo1, be1 + bo1, x1f, x1h);

    launch_gemm(1, 1, x1h, W1p + w1o, b1 + b1o, nullptr, hh, MM, MFF, MD, stream);  // GELU
    launch_gemm(0, 0, hh, W2p + w2o, b2 + bo1, ybf, nullptr, MM, MD, MFF, stream);
    add_ln_kernel<<<dim3(MM), dim3(128), 0, stream>>>(x1f, ybf, g2 + bo1, be2 + bo1, x0f, x0h);
    xcF = x0f; xcH = x0h;
  }

  add_ln_kernel<<<dim3(MM), dim3(128), 0, stream>>>(xcF, nullptr, gf, bef, ybf, xfh);
  launch_gemm(0, 3, xfh, Wlp, bl, out, nullptr, MM, MV, MD, stream);  // NT stream-out
}
